// ConvSeg_71476845740770
// MI455X (gfx1250) — compile-verified, matrix operand order checked by reading, not run on hardware
//
#include <hip/hip_runtime.h>
#include <hip/hip_bf16.h>

#define TOK_N  32768
#define DIN    256
#define DOUT   256
#define KW     9
#define CPAD   4
#define RED    (DIN * KW)      // 2304
#define NCHUNK (RED / 32)      // 72
#define BN_EPS 1e-5f

// ws layout (bytes):
//   [0,      2048)  : 512 f32 BN stats (sum / sumsq)
//   [2048,   2560)  : 256 bf16 zero row (masked-tap A source)
//   [2560,   2560 + 1179648) : W bf16 B-fragments
//   [1182208, +16MB): bf16 copy of x
#define WS_ZROW_OFF  2048
#define WS_WFRAG_OFF 2560
#define WS_XB_OFF    (2560 + (size_t)NCHUNK * 16 * 32 * 16 * 2)  // 1182208

typedef __attribute__((ext_vector_type(16))) __bf16 v16bf;
typedef __attribute__((ext_vector_type(8)))  float  v8f;

union BfFrag {
    v16bf v;
    unsigned short u[16];
    uint4 q[2];
};

// round-to-nearest-even f32 -> bf16 (prep kernels only; not in hot loop)
__device__ __forceinline__ unsigned short f2bf(float f) {
    unsigned int u = __float_as_uint(f);
    u += 0x7FFFu + ((u >> 16) & 1u);
    return (unsigned short)(u >> 16);
}

// ---------------------------------------------------------------------------
// Zero the BN stats (512 f32) and the bf16 zero row (512 B = 128 f32).
// ---------------------------------------------------------------------------
__global__ __launch_bounds__(640) void init_ws_kernel(float* ws) {
    int t = threadIdx.x;
    if (t < 512) {
        ws[t] = 0.0f;                                   // stats
    } else {
        ((float*)((char*)ws + WS_ZROW_OFF))[t - 512] = 0.0f;  // zero row
    }
}

// ---------------------------------------------------------------------------
// Convert x (fp32 [TOK_N][DIN]) to bf16 row-major once, so the WMMA hot loop
// does zero conversion VALU work. 4 elements per thread.
// ---------------------------------------------------------------------------
__global__ __launch_bounds__(256) void prep_x_kernel(
    const float* __restrict__ x, unsigned short* __restrict__ xb)
{
    int idx = (blockIdx.x * 256 + threadIdx.x) * 4;  // < TOK_N*DIN
    float4 f = *(const float4*)(x + idx);
    ushort4 h;
    h.x = f2bf(f.x); h.y = f2bf(f.y); h.z = f2bf(f.z); h.w = f2bf(f.w);
    *(ushort4*)(xb + idx) = h;
}

// ---------------------------------------------------------------------------
// Repack W[o, d, k] (fp32, row-major [DOUT][DIN][KW]) into bf16 B-fragments in
// the exact 16-bit B 32x16 WMMA lane layout:
//   wfrag[chunk][ogrp][lane][e]   (e = 0..15 bf16, 32 bytes per lane)
// chunk c: tap k = c/8, d-range = (c%8)*32 .. +32
// lane l : column N = l&15 -> channel o = ogrp*16 + (l&15)
//          kb = (l<16) ? 0 : 8 ;  e<8 -> K = kb+e ; e>=8 -> K = kb+16+(e-8)
// ---------------------------------------------------------------------------
__global__ __launch_bounds__(256) void prep_w_kernel(
    const float* __restrict__ W, unsigned short* __restrict__ wfrag)
{
    int idx   = blockIdx.x * 256 + threadIdx.x;   // < NCHUNK*16*32*16
    int e     = idx & 15;
    int lane  = (idx >> 4) & 31;
    int ogrp  = (idx >> 9) & 15;
    int chunk = idx >> 13;
    int k     = chunk >> 3;
    int dbase = (chunk & 7) * 32;
    int o     = ogrp * 16 + (lane & 15);
    int kb    = (lane < 16) ? 0 : 8;
    int K     = (e < 8) ? (kb + e) : (kb + 16 + (e - 8));
    int d     = dbase + K;
    wfrag[idx] = f2bf(W[((size_t)o * DIN + d) * KW + k]);
}

// ---------------------------------------------------------------------------
// Implicit-GEMM segment conv + bias + ReLU via v_wmma_f32_16x16x32_bf16.
// WG = 256 threads = 8 waves; wave tile = 16 tokens x 128 channels
// (8 accumulators); WG tile = 64 tokens x 256 channels.
// Outer loop over 9 conv taps selects per-lane A row pointer (real row or
// zero row) ONCE per tap -> the inner 8-chunk loop is completely branch-free:
// 2 b128 A loads + 8 x 32B B loads + 8 WMMA per chunk.
// ---------------------------------------------------------------------------
__global__ __launch_bounds__(256) void conv_wmma_kernel(
    const unsigned short* __restrict__ xb, const unsigned short* __restrict__ wfrag,
    const unsigned short* __restrict__ zrow,
    const float* __restrict__ bias, const int* __restrict__ seg,
    float* __restrict__ out)
{
    const int lane  = threadIdx.x & 31;
    const int wid   = threadIdx.x >> 5;
    const int msub  = wid & 3;        // 4 M-subtiles of 16 tokens
    const int nhalf = wid >> 2;       // 2 halves of 128 channels
    const int mbase = blockIdx.x * 64 + msub * 16;
    const int M     = lane & 15;
    const int kb    = (lane < 16) ? 0 : 8;
    const int n     = mbase + M;      // this lane's output token row
    const int segn  = seg[n];

    const v8f zero = {};
    v8f acc[8];
#pragma unroll
    for (int g = 0; g < 8; ++g) acc[g] = zero;

    for (int k = 0; k < KW; ++k) {
        // Branch-free per-tap validity + pointer select (per-lane cndmask,
        // no exec divergence). Clamped index keeps the seg probe in-bounds.
        const int  src  = n + k - CPAD;
        const int  srcc = min(max(src, 0), TOK_N - 1);
        const bool ok   = (src >= 0) && (src < TOK_N) && (seg[srcc] == segn);
        const unsigned short* xr =
            (ok ? (xb + (size_t)src * DIN) : zrow) + kb;

        const unsigned short* wb =
            wfrag + (((size_t)(k * 8) * 16 + nhalf * 8) * 32 + lane) * 16;

        for (int cc = 0; cc < 8; ++cc) {
            BfFrag a;
            // elems 0..7 : d = cc*32+kb+0..7   (16 contiguous bytes)
            // elems 8..15: d = cc*32+kb+16..23 (16 contiguous bytes)
            const unsigned short* xp = xr + cc * 32;
            a.q[0] = *(const uint4*)(xp + 0);
            a.q[1] = *(const uint4*)(xp + 16);

#pragma unroll
            for (int g = 0; g < 8; ++g) {
                BfFrag b;
                b.v = *(const v16bf*)(wb + (size_t)cc * 8192 + (size_t)g * 512);
                acc[g] = __builtin_amdgcn_wmma_f32_16x16x32_bf16(
                    false, a.v, false, b.v, (short)0, acc[g], false, false);
            }
        }
    }

    // Epilogue: C/D layout -> lanes 0-15: VGPR j holds M=j; lanes 16-31: M=j+8
#pragma unroll
    for (int g = 0; g < 8; ++g) {
        const int   col = nhalf * 128 + g * 16 + M;
        const float bb  = bias[col];
#pragma unroll
        for (int j = 0; j < 8; ++j) {
            int   m = mbase + ((lane < 16) ? j : j + 8);
            float v = acc[g][j] + bb;
            out[(size_t)m * DOUT + col] = (v > 0.f) ? v : 0.f;
        }
    }
}

// ---------------------------------------------------------------------------
// BatchNorm statistics and normalization
// ---------------------------------------------------------------------------
__global__ __launch_bounds__(256) void stats_kernel(
    const float* __restrict__ y, float* __restrict__ ws)
{
    const int c     = threadIdx.x;           // channel
    const int rbase = blockIdx.x * 128;
    float s = 0.f, s2 = 0.f;
    for (int r = 0; r < 128; ++r) {
        float v = y[(size_t)(rbase + r) * DOUT + c];
        s  += v;
        s2 += v * v;
    }
    atomicAdd(&ws[c], s);
    atomicAdd(&ws[DOUT + c], s2);
}

__global__ __launch_bounds__(256) void bn_norm_kernel(
    float* __restrict__ y, const float* __restrict__ ws,
    const float* __restrict__ gamma, const float* __restrict__ beta)
{
    const int idx  = blockIdx.x * 256 + threadIdx.x;  // over TOK_N*DOUT
    const int c    = idx & (DOUT - 1);
    const float im = 1.0f / (float)TOK_N;
    float mean = ws[c] * im;
    float var  = ws[DOUT + c] * im - mean * mean;
    float scl  = rsqrtf(var + BN_EPS) * gamma[c];
    float v    = y[idx];
    y[idx] = (v - mean) * scl + beta[c];
}

// ---------------------------------------------------------------------------
extern "C" void kernel_launch(void* const* d_in, const int* in_sizes, int n_in,
                              void* d_out, int out_size, void* d_ws, size_t ws_size,
                              hipStream_t stream) {
    const float* x     = (const float*)d_in[0];   // [32768, 256] f32
    const float* W     = (const float*)d_in[1];   // [256, 256, 9] f32
    const float* bias  = (const float*)d_in[2];   // [256] f32
    const float* gamma = (const float*)d_in[3];   // [256] f32
    const float* beta  = (const float*)d_in[4];   // [256] f32
    const int*   seg   = (const int*)d_in[5];     // [32768] i32

    float*          out   = (float*)d_out;                               // [32768, 256]
    float*          stats = (float*)d_ws;                                // 512 f32
    unsigned short* zrow  = (unsigned short*)((char*)d_ws + WS_ZROW_OFF);
    unsigned short* wfrag = (unsigned short*)((char*)d_ws + WS_WFRAG_OFF);
    unsigned short* xb    = (unsigned short*)((char*)d_ws + WS_XB_OFF);

    init_ws_kernel<<<1, 640, 0, stream>>>(stats);
    prep_x_kernel<<<(TOK_N * DIN) / (256 * 4), 256, 0, stream>>>(x, xb);
    prep_w_kernel<<<(NCHUNK * 16 * 32 * 16) / 256, 256, 0, stream>>>(W, wfrag);
    conv_wmma_kernel<<<TOK_N / 64, 256, 0, stream>>>(xb, wfrag, zrow, bias, seg, out);
    stats_kernel<<<TOK_N / 128, 256, 0, stream>>>(out, stats);
    bn_norm_kernel<<<(TOK_N * DOUT) / 256, 256, 0, stream>>>(out, stats, gamma, beta);
}
